// ReceptorEncoder_64098091925555
// MI455X (gfx1250) — matrix operand order repaired and checked by reading, not compiled, not run on hardware
//
#include <hip/hip_runtime.h>
#include <hip/hip_bf16.h>
#include <math.h>

typedef __attribute__((ext_vector_type(16))) __bf16 v16bf;
typedef __attribute__((ext_vector_type(8)))  float  v8f;

__device__ __forceinline__ float siluf(float v) { return v / (1.0f + __expf(-v)); }

__device__ __forceinline__ v8f wmma_bf16(v16bf a, v16bf b, v8f c) {
  return __builtin_amdgcn_wmma_f32_16x16x32_bf16(false, a, false, b, (short)0, c, false, false);
}

// A-fragment swizzle (16-bit A 16x32, ISA 7.12.2):
//   kk in [0,32): hh=(kk>>3)&1 selects lane half; vgpr v = 4*(kk>>4) + ((kk>>1)&3); byte-pair j=kk&1
__device__ __forceinline__ int a_pos(int kk) {
  return ((((kk >> 4) & 1) * 4) + ((kk >> 1) & 3)) * 2 + (kk & 1);
}

// ---------------------------------------------------------------------------
// Pack f32 weight [K x 256] into bf16 WMMA B-fragment order:
// buffer index ((t*NC + c)*32 + lane)*16 + pos ; lane = hh*16 + n%16, kk = (pos>>3)*16 + hh*8 + (pos&7)
// ---------------------------------------------------------------------------
__global__ void pack_b_kernel(const float* __restrict__ W, int K, int Nw,
                              __bf16* __restrict__ out, int NC) {
  int total = 16 * NC * 512;
  for (int idx = blockIdx.x * blockDim.x + threadIdx.x; idx < total;
       idx += gridDim.x * blockDim.x) {
    int pos  = idx & 15;
    int lane = (idx >> 4) & 31;
    int c    = (idx >> 9) % NC;
    int t    = idx / (NC * 512);
    int n    = t * 16 + (lane & 15);
    int kk   = ((pos >> 3) * 16) + ((lane >> 4) * 8) + (pos & 7);
    int k    = c * 32 + kk;
    float v  = (k < K && n < Nw) ? W[(size_t)k * Nw + n] : 0.0f;
    out[idx] = (__bf16)v;
  }
}

__global__ void f32_to_bf16_kernel(const float* __restrict__ in,
                                   __bf16* __restrict__ out, int n) {
  for (int i = blockIdx.x * blockDim.x + threadIdx.x; i < n;
       i += gridDim.x * blockDim.x)
    out[i] = (__bf16)in[i];
}

__global__ void deg_kernel(const int* __restrict__ dst, int E, float* __restrict__ deg) {
  int i = blockIdx.x * blockDim.x + threadIdx.x;
  if (i < E) atomicAdd(&deg[dst[i]], 1.0f);
}

__global__ void x_update_kernel(float* __restrict__ x, const float* __restrict__ xn,
                                const float* __restrict__ deg, int n) {
  int i = blockIdx.x * blockDim.x + threadIdx.x;
  if (i < n) {
    float d = fmaxf(deg[i], 1.0f);
    x[i * 3 + 0] += xn[i * 3 + 0] / d;
    x[i * 3 + 1] += xn[i * 3 + 1] / d;
    x[i * 3 + 2] += xn[i * 3 + 2] / d;
  }
}

// ---------------------------------------------------------------------------
// Fused EGNN edge kernel: 64 edges/block, 128 threads = 4 waves, 1 M-tile/wave.
//   A1 = [h_src | h_dst | radial] (bf16, LDS, fragment-swizzled)
//   t   = silu(A1 @ We1 + be1)           (GEMM1, K = NC1*32)
//   m   = silu(t  @ We2 + be2)           (GEMM2, K = 256) -> atomic h_neigh
//   c   = silu(m  @ Wc1 + bc1) @ Wc2     (GEMM3 + dot)    -> tanh(c)*dir*10 -> x_neigh
// ---------------------------------------------------------------------------
__global__ void edge_conv_kernel(
    const __bf16* __restrict__ hb, int fin,
    const float* __restrict__ x,
    const int* __restrict__ src, const int* __restrict__ dst, int E,
    const __bf16* __restrict__ We1p, int NC1, const float* __restrict__ be1,
    const __bf16* __restrict__ We2p, const float* __restrict__ be2,
    const __bf16* __restrict__ Wc1p, const float* __restrict__ bc1,
    const float* __restrict__ Wc2,
    float* __restrict__ h_neigh, float* __restrict__ x_neigh) {
  extern __shared__ char smem[];
  const int tid  = threadIdx.x;
  const int lane = tid & 31;
  const int wv   = tid >> 5;          // M-tile (0..3)
  const int ebase = blockIdx.x * 64;

  __bf16* sA  = (__bf16*)smem;                                  // 4*NC1*512
  __bf16* sT  = (__bf16*)(smem + (size_t)4 * NC1 * 1024);       // 4*8*512
  __bf16* sT2 = sT + 4 * 8 * 512;                               // 4*8*512
  float*  sDir = (float*)(sT2 + 4 * 8 * 512);
  float*  sRad = sDir + 64 * 3;
  float*  sC   = sRad + 64;
  int*    sSrc = (int*)(sC + 64);
  int*    sDst = sSrc + 64;

  if (tid < 64) {
    int e = ebase + tid;
    int s = 0, d = 0;
    float dx = 0.f, dy = 0.f, dz = 0.f, rad = 0.f;
    if (e < E) {
      s = src[e]; d = dst[e];
      dx = x[s * 3 + 0] - x[d * 3 + 0];
      dy = x[s * 3 + 1] - x[d * 3 + 1];
      dz = x[s * 3 + 2] - x[d * 3 + 2];
      rad = dx * dx + dy * dy + dz * dz;
      float inv = 1.0f / (sqrtf(rad) + 1e-30f);
      dx *= inv; dy *= inv; dz *= inv;
      __builtin_prefetch(hb + (size_t)s * fin, 0, 1);
      __builtin_prefetch(hb + (size_t)d * fin, 0, 1);
    }
    sSrc[tid] = s; sDst[tid] = d; sRad[tid] = rad;
    sDir[tid * 3 + 0] = dx; sDir[tid * 3 + 1] = dy; sDir[tid * 3 + 2] = dz;
    sC[tid] = 0.0f;
  }
  __syncthreads();

  // Stage A1 (fragment-swizzled)
  const int K1 = 2 * fin + 1;
  const int Kpad = NC1 * 32;
  for (int idx = tid; idx < 64 * Kpad; idx += 128) {
    int el = idx / Kpad;
    int k  = idx - el * Kpad;
    float v = 0.0f;
    int e = ebase + el;
    if (e < E && k < K1) {
      if (k < fin)            v = (float)hb[(size_t)sSrc[el] * fin + k];
      else if (k < 2 * fin)   v = (float)hb[(size_t)sDst[el] * fin + (k - fin)];
      else                    v = sRad[el];
    }
    int m = el >> 4, r = el & 15;
    int c = k >> 5, kk = k & 31;
    int hh = (kk >> 3) & 1;
    sA[(((m * NC1 + c) * 32) + (hh * 16 + r)) * 16 + a_pos(kk)] = (__bf16)v;
  }
  __syncthreads();

  const int n16  = lane & 15;
  const int half = lane >> 4;

  // GEMM1: silu(A1 @ We1 + be1) -> sT (swizzled as A for GEMM2)
  for (int t = 0; t < 16; ++t) {
    v8f acc = {};
    for (int c = 0; c < NC1; ++c) {
      v16bf a = *(const v16bf*)(sA + (((wv * NC1 + c) * 32) + lane) * 16);
      v16bf b = *(const v16bf*)(We1p + (((t * NC1 + c) * 32) + lane) * 16);
      acc = wmma_bf16(a, b, acc);
    }
    int n = t * 16 + n16;
    float bias = be1[n];
    int c2 = n >> 5, kk = n & 31;
    int hh2 = (kk >> 3) & 1, pos = a_pos(kk);
#pragma unroll
    for (int r = 0; r < 8; ++r) {
      int row = r + half * 8;
      float vv = siluf(acc[r] + bias);
      sT[(((wv * 8 + c2) * 32) + (hh2 * 16 + row)) * 16 + pos] = (__bf16)vv;
    }
  }

  // GEMM2: silu(sT @ We2 + be2) -> sT2 + atomic scatter into h_neigh
  for (int t = 0; t < 16; ++t) {
    v8f acc = {};
    for (int c = 0; c < 8; ++c) {
      v16bf a = *(const v16bf*)(sT + (((wv * 8 + c) * 32) + lane) * 16);
      v16bf b = *(const v16bf*)(We2p + (((t * 8 + c) * 32) + lane) * 16);
      acc = wmma_bf16(a, b, acc);
    }
    int n = t * 16 + n16;
    float bias = be2[n];
    int c2 = n >> 5, kk = n & 31;
    int hh2 = (kk >> 3) & 1, pos = a_pos(kk);
#pragma unroll
    for (int r = 0; r < 8; ++r) {
      int row = r + half * 8;
      float vv = siluf(acc[r] + bias);
      sT2[(((wv * 8 + c2) * 32) + (hh2 * 16 + row)) * 16 + pos] = (__bf16)vv;
      int el = wv * 16 + row;
      if (ebase + el < E)
        atomicAdd(&h_neigh[(size_t)sDst[el] * 256 + n], vv);
    }
  }

  // GEMM3: (silu(sT2 @ Wc1 + bc1)) . Wc2  -> per-edge scalar via LDS atomics
  for (int t = 0; t < 16; ++t) {
    v8f acc = {};
    for (int c = 0; c < 8; ++c) {
      v16bf a = *(const v16bf*)(sT2 + (((wv * 8 + c) * 32) + lane) * 16);
      v16bf b = *(const v16bf*)(Wc1p + (((t * 8 + c) * 32) + lane) * 16);
      acc = wmma_bf16(a, b, acc);
    }
    int n = t * 16 + n16;
    float bias = bc1[n];
    float wn = Wc2[n];
#pragma unroll
    for (int r = 0; r < 8; ++r) {
      int row = r + half * 8;
      atomicAdd(&sC[wv * 16 + row], siluf(acc[r] + bias) * wn);
    }
  }
  __syncthreads();

  if (tid < 64) {
    int e = ebase + tid;
    if (e < E) {
      float coef = tanhf(sC[tid]) * 10.0f;   // COORDS_RANGE, use_tanh
      int d = sDst[tid];
      atomicAdd(&x_neigh[d * 3 + 0], coef * sDir[tid * 3 + 0]);
      atomicAdd(&x_neigh[d * 3 + 1], coef * sDir[tid * 3 + 1]);
      atomicAdd(&x_neigh[d * 3 + 2], coef * sDir[tid * 3 + 2]);
    }
  }
}

// ---------------------------------------------------------------------------
// Generic row MLP: OUT = act(concat(S1,S2) @ Wp + b), 64 rows/block, WMMA.
// ---------------------------------------------------------------------------
__global__ void row_mlp_kernel(
    const __bf16* __restrict__ S1, int w1,
    const __bf16* __restrict__ S2, int w2,
    const __bf16* __restrict__ Wp, int NC, const float* __restrict__ bias,
    int act, int nrows,
    float* __restrict__ outF, __bf16* __restrict__ outB) {
  extern __shared__ char smem[];
  __bf16* sA = (__bf16*)smem;
  const int tid = threadIdx.x, lane = tid & 31, wv = tid >> 5;
  const int rbase = blockIdx.x * 64;
  const int Kpad = NC * 32, Ktot = w1 + w2;

  for (int idx = tid; idx < 64 * Kpad; idx += 128) {
    int rl = idx / Kpad, k = idx - rl * Kpad;
    int row = rbase + rl;
    float v = 0.0f;
    if (row < nrows && k < Ktot)
      v = (k < w1) ? (float)S1[(size_t)row * w1 + k]
                   : (float)S2[(size_t)row * w2 + (k - w1)];
    int m = rl >> 4, r = rl & 15, c = k >> 5, kk = k & 31;
    int hh = (kk >> 3) & 1;
    sA[(((m * NC + c) * 32) + (hh * 16 + r)) * 16 + a_pos(kk)] = (__bf16)v;
  }
  __syncthreads();

  const int n16 = lane & 15, half = lane >> 4;
  for (int t = 0; t < 16; ++t) {
    v8f acc = {};
    for (int c = 0; c < NC; ++c) {
      v16bf a = *(const v16bf*)(sA + (((wv * NC + c) * 32) + lane) * 16);
      v16bf b = *(const v16bf*)(Wp + (((t * NC + c) * 32) + lane) * 16);
      acc = wmma_bf16(a, b, acc);
    }
    int n = t * 16 + n16;
    float bv = bias[n];
#pragma unroll
    for (int r = 0; r < 8; ++r) {
      int row = rbase + wv * 16 + r + half * 8;
      if (row < nrows) {
        float vv = acc[r] + bv;
        if (act) vv = siluf(vv);
        outF[(size_t)row * 256 + n] = vv;
        if (outB) outB[(size_t)row * 256 + n] = (__bf16)vv;
      }
    }
  }
}

// ------------------------- keypoint-attention tail -------------------------

__global__ void colmean_kernel(const float* __restrict__ nf, int nrows,
                               float* __restrict__ mf) {
  int f = threadIdx.x;
  float a = 0.f;
  for (int n = 0; n < nrows; ++n) a += nf[(size_t)n * 256 + f];
  mf[f] = a / (float)nrows;
}

__global__ void qv_kernel(const float* __restrict__ mf, const float* __restrict__ Wq,
                          float* __restrict__ q) {
  int idx = blockIdx.x * blockDim.x + threadIdx.x;   // idx = k*256 + f
  if (idx < 2560) {
    float a = 0.f;
    for (int c = 0; c < 256; ++c) a += mf[c] * Wq[(size_t)c * 2560 + idx];
    q[idx] = a;
  }
}

__global__ void vmat_kernel(const float* __restrict__ Wk, const float* __restrict__ q,
                            float* __restrict__ V) {
  int idx = blockIdx.x * blockDim.x + threadIdx.x;   // idx = c*10 + k
  if (idx < 2560) {
    int c = idx / 10, k = idx % 10;
    float a = 0.f;
    for (int f = 0; f < 256; ++f)
      a += Wk[(size_t)c * 2560 + k * 256 + f] * q[k * 256 + f];
    V[idx] = a;
  }
}

__global__ void logits_kernel(const float* __restrict__ h, const float* __restrict__ V,
                              float* __restrict__ lg, int nrows) {
  int idx = blockIdx.x * blockDim.x + threadIdx.x;
  if (idx < 10 * nrows) {
    int k = idx / nrows, n = idx % nrows;
    float a = 0.f;
    for (int c = 0; c < 256; ++c) a += h[(size_t)n * 256 + c] * V[c * 10 + k];
    lg[(size_t)k * nrows + n] = a * 0.0625f;   // 1/sqrt(256)
  }
}

__device__ __forceinline__ float block_reduce(float v, float* red, bool is_max) {
  int tid = threadIdx.x;
  red[tid] = v; __syncthreads();
  for (int s = 128; s > 0; s >>= 1) {
    if (tid < s) red[tid] = is_max ? fmaxf(red[tid], red[tid + s]) : red[tid] + red[tid + s];
    __syncthreads();
  }
  float r = red[0]; __syncthreads();
  return r;
}

__global__ void kppos_kernel(const float* __restrict__ lg, const float* __restrict__ x,
                             int nrows, float* __restrict__ kpp, float* __restrict__ out) {
  __shared__ float red[256];
  int k = blockIdx.x, tid = threadIdx.x;
  const float* l = lg + (size_t)k * nrows;
  float m = -1e30f;
  for (int n = tid; n < nrows; n += 256) m = fmaxf(m, l[n]);
  m = block_reduce(m, red, true);
  float se = 0.f;
  for (int n = tid; n < nrows; n += 256) se += __expf(l[n] - m);
  float S = block_reduce(se, red, false);
  float p[3] = {0.f, 0.f, 0.f};
  for (int n = tid; n < nrows; n += 256) {
    float w = __expf(l[n] - m);
    p[0] += w * x[n * 3 + 0]; p[1] += w * x[n * 3 + 1]; p[2] += w * x[n * 3 + 2];
  }
  for (int i = 0; i < 3; ++i) {
    float s = block_reduce(p[i], red, false);
    if (tid == 0) { kpp[k * 3 + i] = s / S; out[k * 3 + i] = s / S; }
  }
}

__global__ void kpfeat_kernel(const float* __restrict__ kpp, const float* __restrict__ x0,
                              const float* __restrict__ h, int nrows,
                              float* __restrict__ kf) {
  __shared__ float red[256];
  __shared__ float sw[256];
  int k = blockIdx.x, tid = threadIdx.x;
  float kx = kpp[k * 3 + 0], ky = kpp[k * 3 + 1], kz = kpp[k * 3 + 2];
  float m = -1e30f;
  for (int n = tid; n < nrows; n += 256) {
    float dx = kx - x0[n * 3 + 0], dy = ky - x0[n * 3 + 1], dz = kz - x0[n * 3 + 2];
    m = fmaxf(m, -sqrtf(dx * dx + dy * dy + dz * dz + 1e-12f));
  }
  m = block_reduce(m, red, true);
  float se = 0.f;
  for (int n = tid; n < nrows; n += 256) {
    float dx = kx - x0[n * 3 + 0], dy = ky - x0[n * 3 + 1], dz = kz - x0[n * 3 + 2];
    se += __expf(-sqrtf(dx * dx + dy * dy + dz * dz + 1e-12f) - m);
  }
  float invS = 1.0f / block_reduce(se, red, false);
  float acc = 0.f;
  for (int base = 0; base < nrows; base += 256) {
    int n = base + tid;
    float w = 0.f;
    if (n < nrows) {
      float dx = kx - x0[n * 3 + 0], dy = ky - x0[n * 3 + 1], dz = kz - x0[n * 3 + 2];
      w = __expf(-sqrtf(dx * dx + dy * dy + dz * dz + 1e-12f) - m) * invS;
    }
    sw[tid] = w; __syncthreads();
    int lim = min(256, nrows - base);
    for (int j = 0; j < lim; ++j) acc += sw[j] * h[(size_t)(base + j) * 256 + tid];
    __syncthreads();
  }
  kf[k * 256 + tid] = acc;
}

__global__ void mlp1_kernel(const float* __restrict__ kf, const float* __restrict__ Wm1,
                            const float* __restrict__ bm1, float* __restrict__ m1) {
  int k = blockIdx.x, o = threadIdx.x;  // 512 threads
  float a = bm1[o];
  for (int c = 0; c < 256; ++c) a += kf[k * 256 + c] * Wm1[(size_t)c * 512 + o];
  m1[k * 512 + o] = siluf(a);
}

__global__ void mlp2_kernel(const float* __restrict__ m1, const float* __restrict__ Wm2,
                            const float* __restrict__ bm2, float* __restrict__ out) {
  int k = blockIdx.x, o = threadIdx.x;  // 256 threads
  float a = bm2[o];
  for (int c = 0; c < 512; ++c) a += m1[k * 512 + c] * Wm2[(size_t)c * 256 + o];
  out[30 + k * 256 + o] = siluf(a);     // after 10x3 kp_pos
}

// ---------------------------------------------------------------------------

extern "C" void kernel_launch(void* const* d_in, const int* in_sizes, int n_in,
                              void* d_out, int out_size, void* d_ws, size_t ws_size,
                              hipStream_t stream) {
  (void)in_sizes; (void)n_in; (void)out_size; (void)ws_size;
  const int N = 10000, E = 320000, F = 256, K = 10;
  const float* h0 = (const float*)d_in[0];
  const float* x0 = (const float*)d_in[1];
  const int*   src = (const int*)d_in[2];
  const int*   dst = (const int*)d_in[3];
  const float* Par[74];
  for (int i = 0; i < 74; ++i) Par[i] = (const float*)d_in[4 + i];
  // layer l leaves: base=l*11 -> We1,be1,We2,be2,Wn1,bn1,Wn2,bn2,Wc1,bc1,Wc2
  // then 66:W_nf 67:b_nf 68:W_q 69:W_k 70:Wm1 71:bm1 72:Wm2 73:bm2

  char* wp = (char*)d_ws;
  auto alloc = [&](size_t b) -> void* {
    void* p = (void*)wp; wp += (b + 255) & ~(size_t)255; return p;
  };
  __bf16* hb  = (__bf16*)alloc((size_t)N * F * 2);
  __bf16* hnb = (__bf16*)alloc((size_t)N * F * 2);
  __bf16* t1b = (__bf16*)alloc((size_t)N * F * 2);
  float* hA   = (float*)alloc((size_t)N * F * 4);
  float* hB   = (float*)alloc((size_t)N * F * 4);
  float* hng  = (float*)alloc((size_t)N * F * 4);
  float* t1f  = (float*)alloc((size_t)N * F * 4);
  float* xb   = (float*)alloc((size_t)N * 3 * 4);
  float* xn   = (float*)alloc((size_t)N * 3 * 4);
  float* degb = (float*)alloc((size_t)N * 4);
  float* mf   = (float*)alloc(256 * 4);
  float* qbuf = (float*)alloc(2560 * 4);
  float* Vbuf = (float*)alloc(2560 * 4);
  float* lg   = (float*)alloc((size_t)K * N * 4);
  float* kpp  = (float*)alloc(K * 3 * 4);
  float* kf   = (float*)alloc(K * 256 * 4);
  float* m1   = (float*)alloc(K * 512 * 4);

  __bf16 *We1p[6], *We2p[6], *Wc1p[6], *Wn1p[6], *Wn2p[6];
  int NC1s[6], NCns[6];
  for (int l = 0; l < 6; ++l) {
    int fin = l ? 256 : 13;
    NC1s[l] = (2 * fin + 1 + 31) / 32;
    NCns[l] = (fin + 256 + 31) / 32;
    We1p[l] = (__bf16*)alloc((size_t)16 * NC1s[l] * 512 * 2);
    We2p[l] = (__bf16*)alloc((size_t)16 * 8 * 512 * 2);
    Wc1p[l] = (__bf16*)alloc((size_t)16 * 8 * 512 * 2);
    Wn1p[l] = (__bf16*)alloc((size_t)16 * NCns[l] * 512 * 2);
    Wn2p[l] = (__bf16*)alloc((size_t)16 * 8 * 512 * 2);
  }
  __bf16* Wnfp = (__bf16*)alloc((size_t)16 * 8 * 512 * 2);

  auto packW = [&](const float* Wm, int Kd, __bf16* out) {
    int NC = (Kd + 31) / 32;
    int tot = 16 * NC * 512;
    pack_b_kernel<<<(tot + 255) / 256, 256, 0, stream>>>(Wm, Kd, 256, out, NC);
  };
  for (int l = 0; l < 6; ++l) {
    int fin = l ? 256 : 13, b = l * 11;
    packW(Par[b + 0], 2 * fin + 1, We1p[l]);
    packW(Par[b + 2], 256, We2p[l]);
    packW(Par[b + 8], 256, Wc1p[l]);
    packW(Par[b + 4], fin + 256, Wn1p[l]);
    packW(Par[b + 6], 256, Wn2p[l]);
  }
  packW(Par[66], 256, Wnfp);

  hipMemsetAsync(degb, 0, (size_t)N * 4, stream);
  deg_kernel<<<(E + 255) / 256, 256, 0, stream>>>(dst, E, degb);
  hipMemcpyAsync(xb, x0, (size_t)N * 3 * 4, hipMemcpyDeviceToDevice, stream);

  const float* hcur = h0;
  float* hnext = hA;
  for (int l = 0; l < 6; ++l) {
    int fin = l ? 256 : 13, b = l * 11;
    int NC1 = NC1s[l], NCn = NCns[l];
    f32_to_bf16_kernel<<<((N * fin) + 255) / 256, 256, 0, stream>>>(hcur, hb, N * fin);
    hipMemsetAsync(hng, 0, (size_t)N * F * 4, stream);
    hipMemsetAsync(xn, 0, (size_t)N * 3 * 4, stream);
    size_t sh1 = (size_t)4 * NC1 * 1024 + 2 * 32768 + 64 * 5 * 4 + 64 * 2 * 4;
    edge_conv_kernel<<<(E + 63) / 64, 128, sh1, stream>>>(
        hb, fin, xb, src, dst, E,
        We1p[l], NC1, Par[b + 1], We2p[l], Par[b + 3],
        Wc1p[l], Par[b + 9], Par[b + 10], hng, xn);
    f32_to_bf16_kernel<<<((N * F) + 255) / 256, 256, 0, stream>>>(hng, hnb, N * F);
    row_mlp_kernel<<<(N + 63) / 64, 128, (size_t)4 * NCn * 1024, stream>>>(
        hb, fin, hnb, 256, Wn1p[l], NCn, Par[b + 5], 1, N, t1f, t1b);
    row_mlp_kernel<<<(N + 63) / 64, 128, (size_t)4 * 8 * 1024, stream>>>(
        t1b, 256, (const __bf16*)nullptr, 0, Wn2p[l], 8, Par[b + 7], 0, N,
        hnext, (__bf16*)nullptr);
    x_update_kernel<<<(N + 255) / 256, 256, 0, stream>>>(xb, xn, degb, N);
    hcur = hnext;
    hnext = (hnext == hA) ? hB : hA;
  }
  const float* hf = hcur;

  f32_to_bf16_kernel<<<((N * F) + 255) / 256, 256, 0, stream>>>(hf, hb, N * F);
  row_mlp_kernel<<<(N + 63) / 64, 128, (size_t)4 * 8 * 1024, stream>>>(
      hb, 256, (const __bf16*)nullptr, 0, Wnfp, 8, Par[67], 1, N, t1f, (__bf16*)nullptr);
  colmean_kernel<<<1, 256, 0, stream>>>(t1f, N, mf);
  qv_kernel<<<10, 256, 0, stream>>>(mf, Par[68], qbuf);
  vmat_kernel<<<10, 256, 0, stream>>>(Par[69], qbuf, Vbuf);
  logits_kernel<<<(10 * N + 255) / 256, 256, 0, stream>>>(hf, Vbuf, lg, N);
  kppos_kernel<<<10, 256, 0, stream>>>(lg, xb, N, kpp, (float*)d_out);
  kpfeat_kernel<<<10, 256, 0, stream>>>(kpp, x0, hf, N, kf);
  mlp1_kernel<<<10, 512, 0, stream>>>(kf, Par[70], Par[71], m1);
  mlp2_kernel<<<10, 256, 0, stream>>>(m1, Par[72], Par[73], (float*)d_out);
}
